// GatLayer_22127671509491
// MI455X (gfx1250) — matrix operand order repaired and checked by reading, not compiled, not run on hardware
//
#include <hip/hip_runtime.h>
#include <math.h>

typedef __attribute__((ext_vector_type(16))) _Float16 v16h;
typedef __attribute__((ext_vector_type(8)))  _Float16 v8h;
typedef __attribute__((ext_vector_type(8)))  float    v8f;

#define IN_FEAT 256
#define N_FEAT  64
#define KC      64     // K-chunk per LDS stage
#define M_TILE  64     // rows per workgroup (4 waves x 16 rows)
#define LSTRIDE 72     // padded LDS row stride in halves (bank-conflict free)

// Build a v16h WMMA fragment from two 16B LDS loads.
__device__ __forceinline__ v16h ld_frag(const _Float16* p0, const _Float16* p1) {
  const v8h a = *(const v8h*)p0;
  const v8h b = *(const v8h*)p1;
  return __builtin_shufflevector(a, b, 0,1,2,3,4,5,6,7,8,9,10,11,12,13,14,15);
}

// gfx1250 async global->LDS copy (ASYNCcnt-tracked, no VGPR round-trip).
// Flat address of a __shared__ object: low 32 bits are the LDS byte offset.
__device__ __forceinline__ void async_copy_b128(_Float16* ldsDst, const _Float16* g) {
  const unsigned lds32 = (unsigned)(size_t)ldsDst;
  asm volatile("global_load_async_to_lds_b128 %0, %1, off"
               :: "v"(lds32), "v"((unsigned long long)(size_t)g) : "memory");
}
__device__ __forceinline__ void wait_asynccnt0() {
  asm volatile("s_wait_asynccnt 0x0" ::: "memory");
}

// ---------------------------------------------------------------------------
// Kernel 1: h = x @ W in fp32 (exact), stored TRANSPOSED as f16: hT[c][i].
// ---------------------------------------------------------------------------
__global__ __launch_bounds__(256)
void hT_kernel(const float* __restrict__ x, const float* __restrict__ W,
               _Float16* __restrict__ hT, int n) {
  __shared__ float xs[4][IN_FEAT];
  const int t   = threadIdx.x;
  const int sub = t >> 6;     // 0..3 : row within block
  const int c   = t & 63;     // output feature
  const int i   = blockIdx.x * 4 + sub;
  const float* xr = x + (size_t)i * IN_FEAT;
  #pragma unroll
  for (int k = c; k < IN_FEAT; k += 64) xs[sub][k] = xr[k];
  __syncthreads();
  float acc = 0.0f;
  #pragma unroll 8
  for (int k = 0; k < IN_FEAT; ++k)
    acc = fmaf(xs[sub][k], W[(k << 6) + c], acc);
  hT[(size_t)c * n + i] = (_Float16)acc;
}

// ---------------------------------------------------------------------------
// Kernel 2: out = elu( (adj>0) @ h / deg )  via f16 WMMA, f32 accumulate.
// adj: register-pipelined clause loads + int->f16 convert + LDS stage.
// hT : double-buffered GLOBAL_LOAD_ASYNC_TO_LDS_B128 (pure copy, ASYNCcnt).
// ---------------------------------------------------------------------------
__global__ __launch_bounds__(128)
void gat_aggregate_kernel(const int* __restrict__ adj,
                          const _Float16* __restrict__ hT,
                          float* __restrict__ out, int n) {
  __shared__ __align__(16) _Float16 As[M_TILE * LSTRIDE];      // adj tile (f16)
  __shared__ __align__(16) _Float16 Bs[2][N_FEAT * LSTRIDE];   // h tile, 2 bufs
  __shared__ unsigned degLDS[M_TILE];

  const int t      = threadIdx.x;
  const int wv     = t >> 5;        // wave 0..3
  const int lane   = t & 31;
  const int lane16 = lane & 15;
  const int hi     = lane >> 4;     // half-wave select (ISA A/B lane layout)
  const size_t m0  = (size_t)blockIdx.x * M_TILE;

  // cooperative-loader mappings
  const int arow  = t >> 4;         // 0..7   (adj: 16 threads per row)
  const int akoff = (t & 15) << 2;  // int offset 0..60 (int4 per thread)
  const int brow  = t >> 3;         // 0..15  (hT: 8 threads per row)
  const int bkoff = (t & 7) << 3;   // half offset 0..56 (16B per thread)

  int degAcc[8];
  #pragma unroll
  for (int p = 0; p < 8; ++p) degAcc[p] = 0;

  v8f acc[4];
  #pragma unroll
  for (int nt = 0; nt < 4; ++nt) acc[nt] = (v8f){0,0,0,0,0,0,0,0};

  const int nChunks = n / KC;

  // ---- prologue: chunk 0 adj into registers; chunk 0 hT async into Bs[0]
  int4 av[8];
  #pragma unroll
  for (int p = 0; p < 8; ++p)
    av[p] = *(const int4*)(adj + (m0 + (p << 3) + arow) * (size_t)n + akoff);
  #pragma unroll
  for (int p = 0; p < 4; ++p) {
    const int rN = (p << 4) + brow;
    async_copy_b128(&Bs[0][rN * LSTRIDE + bkoff], hT + (size_t)rN * n + bkoff);
  }

  for (int ch = 0; ch < nChunks; ++ch) {
    // ---- stage current adj registers into LDS; degree via popcount
    #pragma unroll
    for (int p = 0; p < 8; ++p) {
      const int4 v = av[p];
      const int r = (p << 3) + arow;
      const unsigned long long e0 = (v.x > 0) ? 0x3C00ull : 0ull;  // f16 1.0
      const unsigned long long e1 = (v.y > 0) ? 0x3C00ull : 0ull;
      const unsigned long long e2 = (v.z > 0) ? 0x3C00ull : 0ull;
      const unsigned long long e3 = (v.w > 0) ? 0x3C00ull : 0ull;
      const unsigned long long pk = e0 | (e1 << 16) | (e2 << 32) | (e3 << 48);
      degAcc[p] += (int)__popcll(pk & 0x0400040004000400ull);  // bit10 of 0x3C00
      *(unsigned long long*)(As + r * LSTRIDE + akoff) = pk;
    }
    wait_asynccnt0();   // this wave's Bs[ch&1] async writes done
    __syncthreads();    // publish As stores + everyone's Bs writes

    // ---- issue chunk ch+1: adj clause loads to regs, hT async to other buf
    const int kn = (ch + 1 < nChunks) ? (ch + 1) * KC : 0;
    #pragma unroll
    for (int p = 0; p < 8; ++p)
      av[p] = *(const int4*)(adj + (m0 + (p << 3) + arow) * (size_t)n + kn + akoff);
    #pragma unroll
    for (int p = 0; p < 4; ++p) {
      const int rN = (p << 4) + brow;
      async_copy_b128(&Bs[(ch + 1) & 1][rN * LSTRIDE + bkoff],
                      hT + (size_t)rN * n + kn + bkoff);
    }
    // distance-2 unconditional prefetch of this thread's adj stream
    __builtin_prefetch(adj + (m0 + arow) * (size_t)n +
                       (size_t)(ch + 2) * KC + akoff, 0, 3);

    // ---- LDS fragment loads + 8 WMMAs
    const _Float16* BsCur = Bs[ch & 1];
    const _Float16* aBase = As + (wv * 16 + lane16) * LSTRIDE + hi * 8;
    const v16h A0 = ld_frag(aBase,      aBase + 16);   // K 0..31
    const v16h A1 = ld_frag(aBase + 32, aBase + 48);   // K 32..63
    v16h Bf[8];
    #pragma unroll
    for (int ks = 0; ks < 2; ++ks) {
      #pragma unroll
      for (int nt = 0; nt < 4; ++nt) {
        const _Float16* bp = BsCur + (nt * 16 + lane16) * LSTRIDE + (ks << 5) + hi * 8;
        Bf[ks * 4 + nt] = ld_frag(bp, bp + 16);
      }
    }
    #pragma unroll
    for (int nt = 0; nt < 4; ++nt)
      acc[nt] = __builtin_amdgcn_wmma_f32_16x16x32_f16(
          false, A0, false, Bf[nt], (short)0, acc[nt], false, false);
    #pragma unroll
    for (int nt = 0; nt < 4; ++nt)
      acc[nt] = __builtin_amdgcn_wmma_f32_16x16x32_f16(
          false, A1, false, Bf[4 + nt], (short)0, acc[nt], false, false);
    __syncthreads();    // protect As / Bs[ch&1] before next overwrite
  }
  // (final dangling async into a dead buffer is retired by S_ENDPGM wait-idle)

  // ---- reduce degree partials across the 16 loader threads of each row
  #pragma unroll
  for (int p = 0; p < 8; ++p) {
    int s = degAcc[p];
    #pragma unroll
    for (int off = 8; off; off >>= 1) s += __shfl_xor(s, off, 16);
    if ((t & 15) == 0) degLDS[(p << 3) + arow] = (unsigned)s;
  }
  __syncthreads();

  // ---- epilogue: softmax over row-constant scores == 1/deg; then ELU.
  #pragma unroll
  for (int v = 0; v < 8; ++v) {
    const int row  = wv * 16 + hi * 8 + v;  // C-matrix VGPR/lane -> (M,N) map
    const float dg   = (float)degLDS[row];
    const float rdeg = 1.0f / fmaxf(dg, 1.0f);
    #pragma unroll
    for (int nt = 0; nt < 4; ++nt) {
      const float s = acc[nt][v] * rdeg;
      out[(m0 + row) * N_FEAT + nt * 16 + lane16] = (s > 0.0f) ? s : expm1f(s);
    }
  }
}

// ---------------------------------------------------------------------------
extern "C" void kernel_launch(void* const* d_in, const int* in_sizes, int n_in,
                              void* d_out, int out_size, void* d_ws, size_t ws_size,
                              hipStream_t stream) {
  const int*   adj = (const int*)d_in[0];
  const float* x   = (const float*)d_in[1];
  const float* W   = (const float*)d_in[2];
  // d_in[3] == a : intentionally unused. The attention score e[i] is constant
  // across softmax row i, so softmax reduces exactly to adj[i,j]/deg(i); the
  // 'a' vector and leaky_relu cancel out of the final result.
  float* out = (float*)d_out;
  const int n = in_sizes[1] / IN_FEAT;   // 12288
  _Float16* hT = (_Float16*)d_ws;        // 64 * n halves = 1.5 MB scratch

  hT_kernel<<<n / 4, 256, 0, stream>>>(x, W, hT, n);
  gat_aggregate_kernel<<<n / M_TILE, 128, 0, stream>>>(adj, hT, out, n);
}
